// GTLayer_84722524880938
// MI455X (gfx1250) — matrix operand order repaired and checked by reading, not compiled, not run on hardware
//
#include <hip/hip_runtime.h>
#include <hip/hip_bf16.h>

typedef __attribute__((ext_vector_type(16))) __bf16 v16bf;
typedef __attribute__((ext_vector_type(8)))  float  v8f;

union Frag16 {
    v16bf v;
    uint4 u[2];
};

static constexpr int Bsz  = 2;
static constexpr int Lseq = 2048;
static constexpr int Dm   = 1024;
static constexpr int FFNd = 4096;
static constexpr int NH   = 8;
static constexpr int HD   = 128;   // Dm / NH

// ---------------------------------------------------------------------------
// CDNA5 async global->LDS copy (ASYNCcnt path). VDST = LDS byte address
// (low 32 bits of the generic LDS pointer, per ISA 10.2 aperture rules).
// ---------------------------------------------------------------------------
__device__ __forceinline__ void async_copy_b128(unsigned lds_addr, const void* gaddr)
{
    asm volatile("global_load_async_to_lds_b128 %0, %1, off"
                 :: "v"(lds_addr), "v"((unsigned long long)(size_t)gaddr)
                 : "memory");
}
__device__ __forceinline__ void wait_async0()
{
    asm volatile("s_wait_asynccnt 0x0" ::: "memory");
}

// ---------------------------------------------------------------------------
// elementwise fp32 -> bf16 (8 per thread)
// ---------------------------------------------------------------------------
__global__ __launch_bounds__(256) void cvt_bf16_kernel(
    const float* __restrict__ x, __bf16* __restrict__ y, long n)
{
    long i = ((long)blockIdx.x * 256 + threadIdx.x) * 8;
    if (i >= n) return;
    float4 a = *reinterpret_cast<const float4*>(x + i);
    float4 b = *reinterpret_cast<const float4*>(x + i + 4);
    union { __bf16 h[8]; uint4 u; } o;
    o.h[0] = (__bf16)a.x; o.h[1] = (__bf16)a.y; o.h[2] = (__bf16)a.z; o.h[3] = (__bf16)a.w;
    o.h[4] = (__bf16)b.x; o.h[5] = (__bf16)b.y; o.h[6] = (__bf16)b.z; o.h[7] = (__bf16)b.w;
    *reinterpret_cast<uint4*>(y + i) = o.u;
}

// ---------------------------------------------------------------------------
// Convert W[K][N] fp32 into WMMA B-fragment tiling, bf16:
//   tile (kt, nt) covers K rows [kt*32,+32) x N cols [nt*16,+16).
//   Lane l holds 16 contiguous bf16: rows kt*32 + (l>=16?16:0) + j, col nt*16+(l&15).
//   Fragment load in GEMM = 32 contiguous bytes per lane (1 KB per tile).
// ---------------------------------------------------------------------------
__global__ __launch_bounds__(256) void tile_weights_kernel(
    const float* __restrict__ W, __bf16* __restrict__ Wt, int K, int N)
{
    __shared__ __align__(16) __bf16 Ls[32][128];
    const int t  = threadIdx.x;
    const int kt = blockIdx.x;            // K/32 slabs
    const int n0 = blockIdx.y * 128;      // N/128 slabs
#pragma unroll
    for (int it = 0; it < 4; ++it) {
        int f  = t + it * 256;            // 1024 float4 = 32x128 floats
        int r  = f >> 5;
        int c4 = (f & 31) << 2;
        float4 v = *reinterpret_cast<const float4*>(&W[((long)kt * 32 + r) * N + n0 + c4]);
        __bf16* d = &Ls[r][c4];
        d[0] = (__bf16)v.x; d[1] = (__bf16)v.y;
        d[2] = (__bf16)v.z; d[3] = (__bf16)v.w;
    }
    __syncthreads();
    const int tl = t >> 5, lane = t & 31;
    const int l16 = lane & 15, kh = (lane >> 4) << 4;
    union { __bf16 h[16]; uint4 u[2]; } o;
#pragma unroll
    for (int j = 0; j < 16; ++j) o.h[j] = Ls[kh + j][tl * 16 + l16];
    long tile_id = (long)kt * (N >> 4) + (n0 >> 4) + tl;
    __bf16* dst = Wt + tile_id * 512 + lane * 16;
    *reinterpret_cast<uint4*>(dst)     = o.u[0];
    *reinterpret_cast<uint4*>(dst + 8) = o.u[1];
}

// ---------------------------------------------------------------------------
// bf16-WMMA GEMM: C[M,N] = A[M,K] @ B[K,N]
//   A: bf16 row-major, async double-buffered into LDS (K-step 64)
//   Bt: bf16 fragment-tiled (L2-resident, no LDS)
// EPI: 0 none, 1 bias+relu, 2 bias.   OBF: 1 -> bf16 output, 0 -> fp32.
// ---------------------------------------------------------------------------
template<int EPI, int OBF>
__global__ __launch_bounds__(256) void gemm_bf16_wmma(
    const __bf16* __restrict__ A, const __bf16* __restrict__ Bt,
    const float* __restrict__ bias, void* __restrict__ Cout,
    int M, int N, int K)
{
    __shared__ __align__(16) __bf16 As[2][128][64];   // 32 KB ping-pong
    const int t    = threadIdx.x;
    const int wave = t >> 5;
    const int lane = t & 31;
    const int half = lane >> 4;
    const int l16  = lane & 15;
    const long bm  = (long)blockIdx.y * 128;
    const long bn  = (long)blockIdx.x * 128;
    const int  wm  = (wave & 3) * 32;
    const int  wn  = (wave >> 2) * 64;
    const int  ntiles = N >> 4;
    const int  ntile0 = (int)(bn >> 4) + (wn >> 4);

    auto stageA = [&](int k0, int buf) {
#pragma unroll
        for (int it = 0; it < 4; ++it) {
            int c  = t + it * 256;           // 1024 16-byte chunks
            int r  = c >> 3;
            int cc = (c & 7) << 3;
            async_copy_b128((unsigned)(size_t)&As[buf][r][cc],
                            A + (bm + r) * K + k0 + cc);
        }
    };

    v8f acc[2][4];
#pragma unroll
    for (int i = 0; i < 2; ++i)
#pragma unroll
        for (int j = 0; j < 4; ++j) acc[i][j] = {};

    stageA(0, 0);
    wait_async0();
    __syncthreads();

    for (int k0 = 0; k0 < K; k0 += 64) {
        const int buf = (k0 >> 6) & 1;
        if (k0 + 64 < K) stageA(k0 + 64, buf ^ 1);   // overlap DMA with WMMA

#pragma unroll
        for (int s = 0; s < 2; ++s) {
            const long ktile = (long)((k0 >> 5) + s);
            Frag16 bfrag[4];
#pragma unroll
            for (int nt = 0; nt < 4; ++nt) {
                const __bf16* bp = Bt + (ktile * ntiles + ntile0 + nt) * 512 + lane * 16;
                bfrag[nt].u[0] = *reinterpret_cast<const uint4*>(bp);
                bfrag[nt].u[1] = *reinterpret_cast<const uint4*>(bp + 8);
            }
            Frag16 af[2];
#pragma unroll
            for (int mt = 0; mt < 2; ++mt) {
                const __bf16* ap = &As[buf][wm + mt * 16 + l16][s * 32 + half * 16];
                af[mt].u[0] = *reinterpret_cast<const uint4*>(ap);
                af[mt].u[1] = *reinterpret_cast<const uint4*>(ap + 8);
            }
#pragma unroll
            for (int mt = 0; mt < 2; ++mt)
#pragma unroll
                for (int nt = 0; nt < 4; ++nt)
                    acc[mt][nt] = __builtin_amdgcn_wmma_f32_16x16x32_bf16(
                        false, af[mt].v, false, bfrag[nt].v, (short)0,
                        acc[mt][nt], false, false);
        }
        wait_async0();
        __syncthreads();
    }

    // ---- epilogue: C layout row = i + 8*half, col = l16 ----
#pragma unroll
    for (int mt = 0; mt < 2; ++mt) {
        long row0 = bm + wm + mt * 16 + half * 8;
#pragma unroll
        for (int nt = 0; nt < 4; ++nt) {
            long col = bn + wn + nt * 16 + l16;
            float badd = 0.0f;
            if constexpr (EPI != 0) badd = bias[col];
#pragma unroll
            for (int i = 0; i < 8; ++i) {
                float x = acc[mt][nt][i] + badd;
                if constexpr (EPI == 1) x = fmaxf(x, 0.0f);
                if constexpr (OBF)
                    ((__bf16*)Cout)[(row0 + i) * N + col] = (__bf16)x;
                else
                    ((float*)Cout)[(row0 + i) * N + col] = x;
            }
        }
    }
}

// ---------------------------------------------------------------------------
// Flash attention over bf16 q/k/v, relative-position bias, bf16 output.
// One block = (batch, head, 128-query tile), 8 waves x 16 q-rows.
// Dynamic LDS: Qs 32K + 2xKs 32K + 2xVs(T) 32K + Ps 16K = 112 KB.
// K/V tiles are double-buffered; K staged via async DMA.
// ---------------------------------------------------------------------------
__global__ __launch_bounds__(256) void attn_wmma(
    const __bf16* __restrict__ Qg, const __bf16* __restrict__ Kg,
    const __bf16* __restrict__ Vg,
    const float* __restrict__ rq, const float* __restrict__ rk,
    __bf16* __restrict__ Og)
{
    extern __shared__ __align__(16) char smem[];
    __bf16* Qs  = reinterpret_cast<__bf16*>(smem);  // [128][128]
    __bf16* Ks0 = Qs + 128 * 128;                   // [2][64][128]
    __bf16* Vs0 = Ks0 + 2 * 64 * 128;               // [2][128][64] (V transposed)
    __bf16* Ps  = Vs0 + 2 * 128 * 64;               // [8][16][64]

    const int t    = threadIdx.x;
    const int wave = t >> 5;
    const int lane = t & 31;
    const int half = lane >> 4;
    const int l16  = lane & 15;

    const int qt   = blockIdx.x & 15;
    const int head = (blockIdx.x >> 4) & 7;
    const int b    = blockIdx.x >> 7;
    const long rowbase = (long)b * Lseq;
    const int  q0  = qt * 128;

    auto stageK = [&](int kt, int buf) {
#pragma unroll
        for (int it = 0; it < 4; ++it) {
            int c  = t + it * 256;
            int r  = c >> 4;
            int cc = (c & 15) << 3;
            async_copy_b128((unsigned)(size_t)&Ks0[buf * 8192 + r * 128 + cc],
                            Kg + (rowbase + kt + r) * Dm + head * HD + cc);
        }
    };
    auto stageV = [&](int kt, int buf) {
#pragma unroll
        for (int it = 0; it < 16; ++it) {
            int f  = t + it * 256;           // 4096 uint chunks = 64x64
            int r  = f >> 6;                 // key row
            int cu = f & 63;                 // dim pair
            union { unsigned u; __bf16 h[2]; } cv;
            cv.u = *reinterpret_cast<const unsigned*>(
                &Vg[(rowbase + kt + r) * Dm + head * HD + 2 * cu]);
            Vs0[buf * 8192 + (2 * cu + 0) * 64 + r] = cv.h[0];
            Vs0[buf * 8192 + (2 * cu + 1) * 64 + r] = cv.h[1];
        }
    };

    // ---- async-stage Q tile (128x128 bf16 = 32 KB) + first K/V tile ----
#pragma unroll
    for (int it = 0; it < 8; ++it) {
        int c  = t + it * 256;               // 2048 chunks; 16 per row
        int r  = c >> 4;
        int cc = (c & 15) << 3;
        async_copy_b128((unsigned)(size_t)&Qs[r * 128 + cc],
                        Qg + (rowbase + q0 + r) * Dm + head * HD + cc);
    }
    stageK(0, 0);
    stageV(0, 0);
    wait_async0();
    __syncthreads();

    Frag16 aq[4];
#pragma unroll
    for (int ds = 0; ds < 4; ++ds) {
        const __bf16* ap = &Qs[(wave * 16 + l16) * 128 + ds * 32 + half * 16];
        aq[ds].u[0] = *reinterpret_cast<const uint4*>(ap);
        aq[ds].u[1] = *reinterpret_cast<const uint4*>(ap + 8);
    }

    float4 rqv[8];
#pragma unroll
    for (int i = 0; i < 8; ++i)
        rqv[i] = *reinterpret_cast<const float4*>(
            &rq[(rowbase + q0 + wave * 16 + half * 8 + i) * 4]);

    float m_i[8], l_i[8];
    v8f accO[8];
#pragma unroll
    for (int i = 0; i < 8; ++i) { m_i[i] = -1e30f; l_i[i] = 0.0f; }
#pragma unroll
    for (int ot = 0; ot < 8; ++ot) accO[ot] = {};

    const float sscale = 0.08838834764831845f;   // 1/sqrt(128)

    for (int kt = 0; kt < Lseq; kt += 64) {
        const int buf = (kt >> 6) & 1;
        if (kt + 64 < Lseq) {                // overlap next tile with compute
            stageK(kt + 64, buf ^ 1);
            stageV(kt + 64, buf ^ 1);
        }
        const __bf16* Ks = Ks0 + buf * 8192;
        const __bf16* Vs = Vs0 + buf * 8192;

        // ---- S = Q K^T : 16x64 per wave ----
        v8f accS[4];
#pragma unroll
        for (int nt = 0; nt < 4; ++nt) accS[nt] = {};
#pragma unroll
        for (int nt = 0; nt < 4; ++nt) {
#pragma unroll
            for (int ds = 0; ds < 4; ++ds) {
                Frag16 bk;
                const __bf16* bp = &Ks[(nt * 16 + l16) * 128 + ds * 32 + half * 16];
                bk.u[0] = *reinterpret_cast<const uint4*>(bp);
                bk.u[1] = *reinterpret_cast<const uint4*>(bp + 8);
                accS[nt] = __builtin_amdgcn_wmma_f32_16x16x32_bf16(
                    false, aq[ds].v, false, bk.v, (short)0, accS[nt], false, false);
            }
        }

        // ---- scale + relative bias ----
        float4 rkv[4];
#pragma unroll
        for (int nt = 0; nt < 4; ++nt)
            rkv[nt] = *reinterpret_cast<const float4*>(
                &rk[(rowbase + kt + nt * 16 + l16) * 4]);
        float s[4][8];
#pragma unroll
        for (int nt = 0; nt < 4; ++nt)
#pragma unroll
            for (int i = 0; i < 8; ++i)
                s[nt][i] = accS[nt][i] * sscale +
                           0.5f * (rqv[i].x * rkv[nt].x + rqv[i].y * rkv[nt].y +
                                   rqv[i].z * rkv[nt].z + rqv[i].w * rkv[nt].w);

        // ---- online softmax (rows live across 16-lane halves) ----
#pragma unroll
        for (int i = 0; i < 8; ++i) {
            float mloc = fmaxf(fmaxf(s[0][i], s[1][i]), fmaxf(s[2][i], s[3][i]));
#pragma unroll
            for (int xm = 1; xm < 16; xm <<= 1)
                mloc = fmaxf(mloc, __shfl_xor(mloc, xm, 32));
            float m_new = fmaxf(m_i[i], mloc);
            float corr  = __expf(m_i[i] - m_new);
            float ps = 0.0f;
#pragma unroll
            for (int nt = 0; nt < 4; ++nt) {
                float p = __expf(s[nt][i] - m_new);
                s[nt][i] = p;
                ps += p;
            }
#pragma unroll
            for (int xm = 1; xm < 16; xm <<= 1)
                ps += __shfl_xor(ps, xm, 32);
            l_i[i] = l_i[i] * corr + ps;
            m_i[i] = m_new;
#pragma unroll
            for (int ot = 0; ot < 8; ++ot) accO[ot][i] *= corr;
#pragma unroll
            for (int nt = 0; nt < 4; ++nt)
                Ps[(wave * 16 + half * 8 + i) * 64 + nt * 16 + l16] = (__bf16)s[nt][i];
        }

        // ---- O += P V ----
#pragma unroll
        for (int ks = 0; ks < 2; ++ks) {
            Frag16 apf;
            const __bf16* pp = &Ps[(wave * 16 + l16) * 64 + ks * 32 + half * 16];
            apf.u[0] = *reinterpret_cast<const uint4*>(pp);
            apf.u[1] = *reinterpret_cast<const uint4*>(pp + 8);
#pragma unroll
            for (int ot = 0; ot < 8; ++ot) {
                Frag16 bv;
                const __bf16* bp = &Vs[(ot * 16 + l16) * 64 + ks * 32 + half * 16];
                bv.u[0] = *reinterpret_cast<const uint4*>(bp);
                bv.u[1] = *reinterpret_cast<const uint4*>(bp + 8);
                accO[ot] = __builtin_amdgcn_wmma_f32_16x16x32_bf16(
                    false, apf.v, false, bv.v, (short)0, accO[ot], false, false);
            }
        }
        wait_async0();
        __syncthreads();
    }

    float inv[8];
#pragma unroll
    for (int i = 0; i < 8; ++i) inv[i] = 1.0f / l_i[i];
#pragma unroll
    for (int ot = 0; ot < 8; ++ot) {
#pragma unroll
        for (int i = 0; i < 8; ++i) {
            long row = rowbase + q0 + wave * 16 + half * 8 + i;
            Og[row * Dm + head * HD + ot * 16 + l16] = (__bf16)(accO[ot][i] * inv[i]);
        }
    }
}

// ---------------------------------------------------------------------------
// out = LayerNorm(x + y); fp32 always, optional bf16 copy for next GEMM's A.
// ---------------------------------------------------------------------------
__global__ __launch_bounds__(256) void add_ln_kernel(
    const float* __restrict__ x, const float* __restrict__ y,
    const float* __restrict__ gamma, const float* __restrict__ beta,
    float* __restrict__ out, __bf16* __restrict__ outb)
{
    __shared__ float red[16];
    const int  t    = threadIdx.x;
    const long base = (long)blockIdx.x * Dm;
    float4 vx = *reinterpret_cast<const float4*>(&x[base + t * 4]);
    float4 vy = *reinterpret_cast<const float4*>(&y[base + t * 4]);
    float s0 = vx.x + vy.x, s1 = vx.y + vy.y, s2 = vx.z + vy.z, s3 = vx.w + vy.w;
    float sum = s0 + s1 + s2 + s3;
    float sq  = s0 * s0 + s1 * s1 + s2 * s2 + s3 * s3;
#pragma unroll
    for (int xm = 1; xm < 32; xm <<= 1) {
        sum += __shfl_xor(sum, xm, 32);
        sq  += __shfl_xor(sq,  xm, 32);
    }
    const int wave = t >> 5, lane = t & 31;
    if (lane == 0) { red[wave] = sum; red[8 + wave] = sq; }
    __syncthreads();
    float tsum = 0.0f, tsq = 0.0f;
#pragma unroll
    for (int iw = 0; iw < 8; ++iw) { tsum += red[iw]; tsq += red[8 + iw]; }
    float mu   = tsum * (1.0f / Dm);
    float var  = tsq * (1.0f / Dm) - mu * mu;
    float rstd = rsqrtf(var + 1e-5f);
    float4 g  = *reinterpret_cast<const float4*>(&gamma[t * 4]);
    float4 bb = *reinterpret_cast<const float4*>(&beta[t * 4]);
    float4 o;
    o.x = (s0 - mu) * rstd * g.x + bb.x;
    o.y = (s1 - mu) * rstd * g.y + bb.y;
    o.z = (s2 - mu) * rstd * g.z + bb.z;
    o.w = (s3 - mu) * rstd * g.w + bb.w;
    *reinterpret_cast<float4*>(&out[base + t * 4]) = o;
    if (outb) {
        union { __bf16 h[4]; uint2 u; } ob;
        ob.h[0] = (__bf16)o.x; ob.h[1] = (__bf16)o.y;
        ob.h[2] = (__bf16)o.z; ob.h[3] = (__bf16)o.w;
        *reinterpret_cast<uint2*>(&outb[base + t * 4]) = ob.u;
    }
}

// ---------------------------------------------------------------------------
// r_k = rh @ Wrk, r_q = rh @ Wrq   (RL_DIM = 4)
// ---------------------------------------------------------------------------
__global__ __launch_bounds__(256) void rl_proj_kernel(
    const float* __restrict__ rh, const float* __restrict__ Wrk,
    const float* __restrict__ Wrq, float* __restrict__ rk, float* __restrict__ rq)
{
    int idx = blockIdx.x * 256 + threadIdx.x;
    if (idx >= Bsz * Lseq) return;
    float4 r = *reinterpret_cast<const float4*>(&rh[idx * 4]);
    float4 ok, oq;
    ok.x = r.x * Wrk[0] + r.y * Wrk[4] + r.z * Wrk[8]  + r.w * Wrk[12];
    ok.y = r.x * Wrk[1] + r.y * Wrk[5] + r.z * Wrk[9]  + r.w * Wrk[13];
    ok.z = r.x * Wrk[2] + r.y * Wrk[6] + r.z * Wrk[10] + r.w * Wrk[14];
    ok.w = r.x * Wrk[3] + r.y * Wrk[7] + r.z * Wrk[11] + r.w * Wrk[15];
    oq.x = r.x * Wrq[0] + r.y * Wrq[4] + r.z * Wrq[8]  + r.w * Wrq[12];
    oq.y = r.x * Wrq[1] + r.y * Wrq[5] + r.z * Wrq[9]  + r.w * Wrq[13];
    oq.z = r.x * Wrq[2] + r.y * Wrq[6] + r.z * Wrq[10] + r.w * Wrq[14];
    oq.w = r.x * Wrq[3] + r.y * Wrq[7] + r.z * Wrq[11] + r.w * Wrq[15];
    *reinterpret_cast<float4*>(&rk[idx * 4]) = ok;
    *reinterpret_cast<float4*>(&rq[idx * 4]) = oq;
}

// ---------------------------------------------------------------------------
extern "C" void kernel_launch(void* const* d_in, const int* in_sizes, int n_in,
                              void* d_out, int out_size, void* d_ws, size_t ws_size,
                              hipStream_t stream)
{
    const float* h   = (const float*)d_in[0];
    const float* rh  = (const float*)d_in[1];
    const float* Wq  = (const float*)d_in[2];
    const float* Wk  = (const float*)d_in[3];
    const float* Wv  = (const float*)d_in[4];
    const float* Wo  = (const float*)d_in[5];
    const float* Wrk = (const float*)d_in[6];
    const float* Wrq = (const float*)d_in[7];
    const float* W1  = (const float*)d_in[8];
    const float* b1  = (const float*)d_in[9];
    const float* W2  = (const float*)d_in[10];
    const float* b2  = (const float*)d_in[11];
    const float* g1  = (const float*)d_in[12];
    const float* be1 = (const float*)d_in[13];
    const float* g2  = (const float*)d_in[14];
    const float* be2 = (const float*)d_in[15];
    float* out = (float*)d_out;
    (void)in_sizes; (void)n_in; (void)out_size; (void)ws_size;

    const long rows = (long)Bsz * Lseq;          // 4096
    const long S    = rows * Dm;                 // 4 M elements

    char* p = (char*)d_ws;
    auto take = [&p](size_t bytes) {
        char* r = p;
        p += (bytes + 255) & ~(size_t)255;
        return r;
    };
    __bf16* hb   = (__bf16*)take(S * 2);
    __bf16* qb   = (__bf16*)take(S * 2);
    __bf16* kb   = (__bf16*)take(S * 2);
    __bf16* vb   = (__bf16*)take(S * 2);
    __bf16* ctxb = (__bf16*)take(S * 2);
    __bf16* h1b  = (__bf16*)take(S * 2);
    __bf16* mid  = (__bf16*)take(rows * FFNd * 2);
    __bf16* Wqt  = (__bf16*)take((size_t)Dm * Dm * 2);
    __bf16* Wkt  = (__bf16*)take((size_t)Dm * Dm * 2);
    __bf16* Wvt  = (__bf16*)take((size_t)Dm * Dm * 2);
    __bf16* Wot  = (__bf16*)take((size_t)Dm * Dm * 2);
    __bf16* W1t  = (__bf16*)take((size_t)Dm * FFNd * 2);
    __bf16* W2t  = (__bf16*)take((size_t)FFNd * Dm * 2);
    float*  hsa  = (float*)take(S * 4);
    float*  h1f  = (float*)take(S * 4);
    float*  hf   = (float*)take(S * 4);
    float*  rqb  = (float*)take(rows * 4 * 4);
    float*  rkb  = (float*)take(rows * 4 * 4);

    dim3 blk(256);

    // one-time (per launch) precision / layout conversion
    cvt_bf16_kernel<<<(int)(S / 8 / 256), blk, 0, stream>>>(h, hb, S);
    tile_weights_kernel<<<dim3(Dm / 32, Dm / 128),   blk, 0, stream>>>(Wq, Wqt, Dm, Dm);
    tile_weights_kernel<<<dim3(Dm / 32, Dm / 128),   blk, 0, stream>>>(Wk, Wkt, Dm, Dm);
    tile_weights_kernel<<<dim3(Dm / 32, Dm / 128),   blk, 0, stream>>>(Wv, Wvt, Dm, Dm);
    tile_weights_kernel<<<dim3(Dm / 32, Dm / 128),   blk, 0, stream>>>(Wo, Wot, Dm, Dm);
    tile_weights_kernel<<<dim3(Dm / 32, FFNd / 128), blk, 0, stream>>>(W1, W1t, Dm, FFNd);
    tile_weights_kernel<<<dim3(FFNd / 32, Dm / 128), blk, 0, stream>>>(W2, W2t, FFNd, Dm);
    rl_proj_kernel<<<(int)(rows / 256), blk, 0, stream>>>(rh, Wrk, Wrq, rkb, rqb);

    dim3 g_d(Dm / 128, (int)(rows / 128));
    gemm_bf16_wmma<0, 1><<<g_d, blk, 0, stream>>>(hb, Wqt, nullptr, qb, (int)rows, Dm, Dm);
    gemm_bf16_wmma<0, 1><<<g_d, blk, 0, stream>>>(hb, Wkt, nullptr, kb, (int)rows, Dm, Dm);
    gemm_bf16_wmma<0, 1><<<g_d, blk, 0, stream>>>(hb, Wvt, nullptr, vb, (int)rows, Dm, Dm);

    attn_wmma<<<Bsz * NH * (Lseq / 128), blk, 114688, stream>>>(qb, kb, vb, rqb, rkb, ctxb);

    gemm_bf16_wmma<0, 0><<<g_d, blk, 0, stream>>>(ctxb, Wot, nullptr, hsa, (int)rows, Dm, Dm);
    add_ln_kernel<<<(int)rows, blk, 0, stream>>>(hsa, h, g1, be1, h1f, h1b);

    dim3 g_f(FFNd / 128, (int)(rows / 128));
    gemm_bf16_wmma<1, 1><<<g_f, blk, 0, stream>>>(h1b, W1t, b1, mid, (int)rows, FFNd, Dm);
    gemm_bf16_wmma<2, 0><<<g_d, blk, 0, stream>>>(mid, W2t, b2, hf, (int)rows, Dm, FFNd);

    add_ln_kernel<<<(int)rows, blk, 0, stream>>>(h1f, hf, g2, be2, out, nullptr);
}
